// ResBlockBNN_44186623541989
// MI455X (gfx1250) — 16-bit and 8-bit matrix operand order repaired and checked by reading, compiled, not run on hardware
//
#include <hip/hip_runtime.h>
#include <hip/hip_bf16.h>

typedef __attribute__((ext_vector_type(8)))  int     v8i;
typedef __attribute__((ext_vector_type(8)))  float   v8f;
typedef __attribute__((ext_vector_type(16))) __bf16  v16bf;
typedef __attribute__((ext_vector_type(4)))  int     i32x4;
typedef long long i64;

union AFrag   { i64   d[4]; v8i   v; };
union BFrag   { i32x4 q[2]; v8i   v; };
union BF16Frag{ i32x4 q[2]; v16bf v; };

#define EPSF 1e-5f

__device__ __forceinline__ signed char sgn8(float v) {
  return v > 0.f ? (signed char)1 : (v < 0.f ? (signed char)-1 : (signed char)0);
}
__device__ __forceinline__ float sgnf(float v) {
  return v > 0.f ? 1.f : (v < 0.f ? -1.f : 0.f);
}

#define OFF_H      ((size_t)0)
#define OFF_S1     ((size_t)134217728)
#define OFF_S2     (OFF_S1 + (size_t)17842176)
#define OFF_STATS  (OFF_S2 + (size_t)35684352)
#define OFF_XBF    (OFF_STATS + (size_t)8192)
#define OFF_W1S    (OFF_XBF + (size_t)33554432)
#define OFF_W2S    (OFF_W1S + (size_t)294912)
#define OFF_WSK    (OFF_W2S + (size_t)589824)
#define OFF_TEMB   (OFF_WSK + (size_t)65536)

#define ST_MEAN1 0
#define ST_RSTD1 128
#define ST_B2SUM 256
#define ST_B2SQ  512
#define ST_A1    768
#define ST_A2    1024
#define ST_ASK   1280

__global__ void zero_kernel(i32x4* __restrict__ p, long long n4) {
  i32x4 z = (i32x4)0;
  for (long long i = (long long)blockIdx.x * blockDim.x + threadIdx.x;
       i < n4; i += (long long)gridDim.x * blockDim.x)
    p[i] = z;
}

__global__ __launch_bounds__(256) void bn1_stats_kernel(
    const float* __restrict__ x, float* __restrict__ stats) {
  const int c = blockIdx.x;
  float s = 0.f, q = 0.f;
  for (int n = 0; n < 32; ++n) {
    const float* base = x + ((size_t)(n * 128 + c)) * 4096;
    for (int i = threadIdx.x; i < 4096; i += 256) {
      float v = base[i]; s += v; q += v * v;
    }
  }
  __shared__ float ss[256], sq[256];
  ss[threadIdx.x] = s; sq[threadIdx.x] = q; __syncthreads();
  for (int o = 128; o > 0; o >>= 1) {
    if (threadIdx.x < o) { ss[threadIdx.x] += ss[threadIdx.x + o]; sq[threadIdx.x] += sq[threadIdx.x + o]; }
    __syncthreads();
  }
  if (threadIdx.x == 0) {
    float m = ss[0] * (1.f / 131072.f);
    float var = sq[0] * (1.f / 131072.f) - m * m;
    stats[ST_MEAN1 + c] = m;
    stats[ST_RSTD1 + c] = rsqrtf(var + EPSF);
  }
}

__global__ __launch_bounds__(128) void alpha_kernel(
    const float* __restrict__ w1, const float* __restrict__ w2,
    const float* __restrict__ wskip, float* __restrict__ stats) {
  const int o = blockIdx.x;
  __shared__ float red[128];
  float s;
  s = 0.f;
  for (int i = threadIdx.x; i < 1152; i += 128) s += fabsf(w1[(size_t)o * 1152 + i]);
  red[threadIdx.x] = s; __syncthreads();
  for (int k = 64; k > 0; k >>= 1) { if (threadIdx.x < k) red[threadIdx.x] += red[threadIdx.x + k]; __syncthreads(); }
  if (threadIdx.x == 0) stats[ST_A1 + o] = red[0] * (1.f / 1152.f);
  __syncthreads();
  s = 0.f;
  for (int i = threadIdx.x; i < 2304; i += 128) s += fabsf(w2[(size_t)o * 2304 + i]);
  red[threadIdx.x] = s; __syncthreads();
  for (int k = 64; k > 0; k >>= 1) { if (threadIdx.x < k) red[threadIdx.x] += red[threadIdx.x + k]; __syncthreads(); }
  if (threadIdx.x == 0) stats[ST_A2 + o] = red[0] * (1.f / 2304.f);
  __syncthreads();
  s = fabsf(wskip[(size_t)o * 128 + threadIdx.x]);
  red[threadIdx.x] = s; __syncthreads();
  for (int k = 64; k > 0; k >>= 1) { if (threadIdx.x < k) red[threadIdx.x] += red[threadIdx.x + k]; __syncthreads(); }
  if (threadIdx.x == 0) stats[ST_ASK + o] = red[0] * (1.f / 128.f);
}

__global__ void pack_weights_kernel(
    const float* __restrict__ w1, const float* __restrict__ w2,
    const float* __restrict__ wskip, const float* __restrict__ stats,
    signed char* __restrict__ w1s, signed char* __restrict__ w2s,
    __bf16* __restrict__ wskbf) {
  for (long long idx = (long long)blockIdx.x * blockDim.x + threadIdx.x;
       idx < 589824; idx += (long long)gridDim.x * blockDim.x) {
    {
      int s = (int)(idx >> 16), o = (int)((idx >> 8) & 255), c = (int)(idx & 255);
      w2s[idx] = sgn8(w2[(size_t)o * 2304 + c * 9 + s]);
    }
    if (idx < 294912) {
      int s = (int)(idx >> 15), o = (int)((idx >> 7) & 255), c = (int)(idx & 127);
      w1s[idx] = sgn8(w1[(size_t)o * 1152 + c * 9 + s]);
    }
    if (idx < 32768) {
      int c = (int)(idx & 127), o = (int)(idx >> 7);
      wskbf[idx] = (__bf16)(sgnf(wskip[(size_t)o * 128 + c]) * stats[ST_ASK + o]);
    }
  }
}

__global__ __launch_bounds__(256) void temb_kernel(
    const float* __restrict__ t, const float* __restrict__ tw,
    const float* __restrict__ tb, float* __restrict__ temb) {
  const int n = blockIdx.x, o = threadIdx.x;
  float acc = tb[o];
  const float* tr = t + (size_t)n * 512;
  const float* wr = tw + (size_t)o * 512;
  for (int k = 0; k < 512; ++k) acc += tr[k] * wr[k];
  temb[n * 256 + o] = acc;
}

__global__ void pack1_kernel(const float* __restrict__ x,
                             const float* __restrict__ gamma1,
                             const float* __restrict__ beta1,
                             const float* __restrict__ stats,
                             signed char* __restrict__ s1,
                             __bf16* __restrict__ xbf) {
  long long idx = (long long)blockIdx.x * blockDim.x + threadIdx.x;
  if (idx >= 16777216) return;
  int hw = (int)(idx & 4095);
  int c = (int)((idx >> 12) & 127);
  int n = (int)(idx >> 19);
  float v = x[idx];
  float b = gamma1[c] * ((v - stats[ST_MEAN1 + c]) * stats[ST_RSTD1 + c]) + beta1[c];
  int y = hw >> 6, xx = hw & 63;
  s1[((size_t)(n * 66 + y + 1) * 66 + xx + 1) * 128 + c] = sgn8(b);
  xbf[((size_t)((n << 6) + y) * 64 + xx) * 128 + c] = (__bf16)v;
}

__global__ __launch_bounds__(256) void conv1_wmma_kernel(
    const signed char* __restrict__ s1, const signed char* __restrict__ w1s,
    const float* __restrict__ stats, const float* __restrict__ temb,
    float* __restrict__ h) {
  const int row = blockIdx.x;
  const int n = row >> 6, y = row & 63;
  const int wave = threadIdx.x >> 5, lane = threadIdx.x & 31;
  const int waveM = wave >> 1, waveN = wave & 1;
  const int m0 = waveM << 4;
  const int ncol = (blockIdx.y << 6) + (waveN << 5);
  const int half = lane >> 4, mr = lane & 15;
  const int mpix = m0 + mr;

  v8i acc[2] = {};
  for (int ky = 0; ky < 3; ++ky) {
    const signed char* yrow = s1 + (size_t)(n * 66 + y + ky) * (66 * 128);
    for (int kx = 0; kx < 3; ++kx) {
      const signed char* arow = yrow + (size_t)(mpix + kx) * 128;
      const int s = ky * 3 + kx;
#pragma unroll
      for (int kc = 0; kc < 2; ++kc) {
        const signed char* ap = arow + (kc << 6) + (half << 5);
        AFrag A;
        A.d[0] = *(const i64*)(ap);
        A.d[1] = *(const i64*)(ap + 8);
        A.d[2] = *(const i64*)(ap + 16);
        A.d[3] = *(const i64*)(ap + 24);
        const signed char* bbase =
            w1s + ((size_t)s << 15) + (size_t)(ncol + mr) * 128 + (kc << 6) + (half << 5);
#pragma unroll
        for (int nt = 0; nt < 2; ++nt) {
          BFrag B;
          B.q[0] = *(const i32x4*)(bbase + (nt << 4) * 128);
          B.q[1] = *(const i32x4*)(bbase + (nt << 4) * 128 + 16);
          acc[nt] = __builtin_amdgcn_wmma_i32_16x16x64_iu8(
              true, A.v, true, B.v, acc[nt], false, false);
        }
      }
    }
  }
#pragma unroll
  for (int nt = 0; nt < 2; ++nt) {
    const int o = ncol + (nt << 4) + mr;
    const float al = stats[ST_A1 + o];
    const float tv = temb[(n << 8) + o];
#pragma unroll
    for (int v = 0; v < 8; ++v) {
      const int xp = m0 + (half << 3) + v;
      h[(size_t)((row << 6) + xp) * 256 + o] = al * (float)acc[nt][v] + tv;
    }
  }
}

__global__ __launch_bounds__(256) void bn2_stats_kernel(
    const float* __restrict__ h, float* __restrict__ stats) {
  const int c = threadIdx.x;
  const size_t p0 = (size_t)blockIdx.x * 256;
  float s = 0.f, q = 0.f;
  for (int i = 0; i < 256; ++i) {
    float v = h[(p0 + i) * 256 + c]; s += v; q += v * v;
  }
  atomicAdd(&stats[ST_B2SUM + c], s);
  atomicAdd(&stats[ST_B2SQ + c], q);
}

__global__ void pack2_kernel(const float* __restrict__ h,
                             const float* __restrict__ gamma2,
                             const float* __restrict__ beta2,
                             const float* __restrict__ stats,
                             signed char* __restrict__ s2) {
  long long idx = (long long)blockIdx.x * blockDim.x + threadIdx.x;
  if (idx >= 33554432) return;
  int c = (int)(idx & 255);
  long long pix = idx >> 8;
  float m = stats[ST_B2SUM + c] * (1.f / 131072.f);
  float var = stats[ST_B2SQ + c] * (1.f / 131072.f) - m * m;
  float v = h[idx];
  float b = gamma2[c] * ((v - m) * rsqrtf(var + EPSF)) + beta2[c];
  int n = (int)(pix >> 12);
  int rem = (int)(pix & 4095);
  int y = rem >> 6, xx = rem & 63;
  s2[((size_t)(n * 66 + y + 1) * 66 + xx + 1) * 256 + c] = sgn8(b);
}

__global__ __launch_bounds__(256) void conv2_wmma_kernel(
    const signed char* __restrict__ s2, const signed char* __restrict__ w2s,
    const __bf16* __restrict__ xbf, const __bf16* __restrict__ wsk,
    const float* __restrict__ stats, float* __restrict__ out) {
  const int row = blockIdx.x;
  const int n = row >> 6, y = row & 63;
  const int wave = threadIdx.x >> 5, lane = threadIdx.x & 31;
  const int waveM = wave >> 1, waveN = wave & 1;
  const int m0 = waveM << 4;
  const int ncol = (blockIdx.y << 6) + (waveN << 5);
  const int half = lane >> 4, mr = lane & 15;
  const int mpix = m0 + mr;

  v8i acc[2] = {};
  v8f facc[2] = {};

  for (int ky = 0; ky < 3; ++ky) {
    const signed char* yrow = s2 + (size_t)(n * 66 + y + ky) * (66 * 256);
    for (int kx = 0; kx < 3; ++kx) {
      const signed char* arow = yrow + (size_t)(mpix + kx) * 256;
      const int s = ky * 3 + kx;
#pragma unroll
      for (int kc = 0; kc < 4; ++kc) {
        const signed char* ap = arow + (kc << 6) + (half << 5);
        AFrag A;
        A.d[0] = *(const i64*)(ap);
        A.d[1] = *(const i64*)(ap + 8);
        A.d[2] = *(const i64*)(ap + 16);
        A.d[3] = *(const i64*)(ap + 24);
        const signed char* bbase =
            w2s + ((size_t)s << 16) + (size_t)(ncol + mr) * 256 + (kc << 6) + (half << 5);
#pragma unroll
        for (int nt = 0; nt < 2; ++nt) {
          BFrag B;
          B.q[0] = *(const i32x4*)(bbase + (nt << 4) * 256);
          B.q[1] = *(const i32x4*)(bbase + (nt << 4) * 256 + 16);
          acc[nt] = __builtin_amdgcn_wmma_i32_16x16x64_iu8(
              true, A.v, true, B.v, acc[nt], false, false);
        }
      }
    }
  }

  const __bf16* xrow = xbf + (size_t)((row << 6) + mpix) * 128;
#pragma unroll
  for (int kc = 0; kc < 4; ++kc) {
    BF16Frag Abf;
    const __bf16* ap = xrow + (kc << 5) + (half << 3);
    Abf.q[0] = *(const i32x4*)(ap);
    Abf.q[1] = *(const i32x4*)(ap + 16);
    const __bf16* bbase = wsk + (size_t)(ncol + mr) * 128 + (kc << 5) + (half << 3);
#pragma unroll
    for (int nt = 0; nt < 2; ++nt) {
      BF16Frag Bbf;
      Bbf.q[0] = *(const i32x4*)(bbase + (nt << 4) * 128);
      Bbf.q[1] = *(const i32x4*)(bbase + (nt << 4) * 128 + 16);
      facc[nt] = __builtin_amdgcn_wmma_f32_16x16x32_bf16(
          false, Abf.v, false, Bbf.v, (short)0, facc[nt], false, false);
    }
  }

#pragma unroll
  for (int nt = 0; nt < 2; ++nt) {
    const int o = ncol + (nt << 4) + mr;
    const float al = stats[ST_A2 + o];
#pragma unroll
    for (int v = 0; v < 8; ++v) {
      const int xp = m0 + (half << 3) + v;
      out[((size_t)(n * 256 + o) << 12) + (y << 6) + xp] =
          al * (float)acc[nt][v] + facc[nt][v];
    }
  }
}

extern "C" void kernel_launch(void* const* d_in, const int* in_sizes, int n_in,
                              void* d_out, int out_size, void* d_ws, size_t ws_size,
                              hipStream_t stream) {
  const float* x      = (const float*)d_in[0];
  const float* t      = (const float*)d_in[1];
  const float* w1     = (const float*)d_in[2];
  const float* w2     = (const float*)d_in[3];
  const float* wskip  = (const float*)d_in[4];
  const float* gamma1 = (const float*)d_in[5];
  const float* beta1  = (const float*)d_in[6];
  const float* gamma2 = (const float*)d_in[7];
  const float* beta2  = (const float*)d_in[8];
  const float* tw     = (const float*)d_in[9];
  const float* tb     = (const float*)d_in[10];
  float* out = (float*)d_out;

  char* ws = (char*)d_ws;
  float*       h     = (float*)(ws + OFF_H);
  signed char* s1    = (signed char*)(ws + OFF_S1);
  signed char* s2    = (signed char*)(ws + OFF_S2);
  float*       stats = (float*)(ws + OFF_STATS);
  __bf16*      xbf   = (__bf16*)(ws + OFF_XBF);
  signed char* w1s   = (signed char*)(ws + OFF_W1S);
  signed char* w2s   = (signed char*)(ws + OFF_W2S);
  __bf16*      wskbf = (__bf16*)(ws + OFF_WSK);
  float*       temb  = (float*)(ws + OFF_TEMB);

  zero_kernel<<<4096, 256, 0, stream>>>((i32x4*)(ws + OFF_S1), 3345920LL);
  bn1_stats_kernel<<<128, 256, 0, stream>>>(x, stats);
  alpha_kernel<<<256, 128, 0, stream>>>(w1, w2, wskip, stats);
  pack_weights_kernel<<<2304, 256, 0, stream>>>(w1, w2, wskip, stats, w1s, w2s, wskbf);
  temb_kernel<<<32, 256, 0, stream>>>(t, tw, tb, temb);
  pack1_kernel<<<65536, 256, 0, stream>>>(x, gamma1, beta1, stats, s1, xbf);
  conv1_wmma_kernel<<<dim3(2048, 4), 256, 0, stream>>>(s1, w1s, stats, temb, h);
  bn2_stats_kernel<<<512, 256, 0, stream>>>(h, stats);
  pack2_kernel<<<131072, 256, 0, stream>>>(h, gamma2, beta2, stats, s2);
  conv2_wmma_kernel<<<dim3(2048, 4), 256, 0, stream>>>(s2, w2s, xbf, wskbf, stats, out);
}
